// BigBIRDSelfAttention_71674414235759
// MI455X (gfx1250) — matrix operand order repaired and checked by reading, not compiled, not run on hardware
//
#include <hip/hip_runtime.h>

// ---------------------------------------------------------------------------
// BigBird self-attention for MI455X (gfx1250, wave32, WMMA).
// B=4, L=4096, D=768. qkv fp32 [B, L, 3*768]; out fp32 [B, L, 768].
// Sparse flash attention: per-wave 16-row tile, 32-col j-tiles, skipping
// tiles that are fully masked (causal + window(384) + global(256) + random
// token pairs pre-scattered into bitmaps in d_ws).
// Matmuls via v_wmma_f32_16x16x32_f16 (f16 operands, f32 accumulate).
// Round 3: software-pipelined B-fragment loads in QK and PV loops; csum
// WMMA hoisted before the PV loop (rsum update after) to kill the
// WMMA->VALU hazard v_nops.
// ---------------------------------------------------------------------------

typedef _Float16 v16h __attribute__((ext_vector_type(16)));
typedef _Float16 v8h  __attribute__((ext_vector_type(8)));
typedef _Float16 h2   __attribute__((ext_vector_type(2)));
typedef float    v8f  __attribute__((ext_vector_type(8)));

#define L_TOK   4096
#define DMODEL  768
#define D3      2304
#define NBATCH  4
#define WINDOW  384
#define GLOBALC 256
#define KPITCH  776            // 768 + 8 halves row pitch (bank-conflict pad)
#define VPITCH  40             // 32 + 8 halves (80B rows -> 16B aligned)
#define PPITCH  40
#define NKCHUNK 24             // 768 / 32
#define NNCHUNK 48             // 768 / 16
#define SCALE_Q 0.036084391824351615f   // 1/sqrt(768)
#define FINE_WORDS_PER_ROW 128          // 4096/32
#define FINE_WORDS  (L_TOK * FINE_WORDS_PER_ROW)   // 524288 u32 = 2MB
#define COARSE_WORDS (256 * 8)                     // 2048 u32 = 8KB

__device__ __forceinline__ v16h ld_v16h(const _Float16* p) {
    v8h lo = *(const v8h*)(p);
    v8h hi = *(const v8h*)(p + 16);
    return __builtin_shufflevector(lo, hi, 0,1,2,3,4,5,6,7,8,9,10,11,12,13,14,15);
}

__device__ __forceinline__ v8f wmma_f16(v16h a, v16h b, v8f c) {
    // D = A(16x32 f16) * B(32x16 f16) + C(16x16 f32)
    return __builtin_amdgcn_wmma_f32_16x16x32_f16(
        /*neg_a=*/false, a, /*neg_b=*/false, b,
        /*c_mod=*/(short)0, c, /*reuse_a=*/false, /*reuse_b=*/false);
}

// ---- 16-lane max butterfly (independent per 16-lane group) ---------------
#if __has_builtin(__builtin_amdgcn_permlane16)
__device__ __forceinline__ float pl16(float x, unsigned s0, unsigned s1) {
    int xi = __float_as_int(x);
    int r  = __builtin_amdgcn_permlane16(xi, xi, (int)s0, (int)s1, false, false);
    return __int_as_float(r);
}
__device__ __forceinline__ float half16_max(float t) {
    t = fmaxf(t, pl16(t, 0x67452301u, 0xEFCDAB89u));   // xor 1
    t = fmaxf(t, pl16(t, 0x54761032u, 0xDCFE98BAu));   // xor 2
    t = fmaxf(t, pl16(t, 0x32107654u, 0xBA98FEDCu));   // xor 4
    t = fmaxf(t, pl16(t, 0xFEDCBA98u, 0x76543210u));   // xor 8
    return t;
}
#else
__device__ __forceinline__ float half16_max(float t) {
    t = fmaxf(t, __shfl_xor(t, 1));
    t = fmaxf(t, __shfl_xor(t, 2));
    t = fmaxf(t, __shfl_xor(t, 4));
    t = fmaxf(t, __shfl_xor(t, 8));
    return t;
}
#endif

// ---------------------------------------------------------------------------
// Kernel 1: scatter random token pairs into fine (per-token) and coarse
// (16x16 block) bitmaps held in d_ws. Bitmaps must be pre-zeroed.
// ---------------------------------------------------------------------------
__global__ void bb_scatter(const int* __restrict__ pos0,
                           const int* __restrict__ pos1,
                           unsigned* __restrict__ fineBM,
                           unsigned* __restrict__ coarseBM,
                           int n) {
    int t = blockIdx.x * blockDim.x + threadIdx.x;
    if (t >= n) return;
    int p0 = pos0[t], p1 = pos1[t];
    if (p0 <= p1) return;   // invalid pairs are no-ops (ref redirects to (0,0))
    #pragma unroll
    for (int di = 0; di < 2; ++di) {
        #pragma unroll
        for (int dj = 0; dj < 2; ++dj) {
            int i = p0 + di;   // <= 4095 (p0 <= 4094)
            int j = p1 + dj;
            atomicOr(&fineBM[i * FINE_WORDS_PER_ROW + (j >> 5)], 1u << (j & 31));
            atomicOr(&coarseBM[(i >> 4) * 8 + (j >> 9)], 1u << ((j >> 4) & 31));
        }
    }
}

// ---------------------------------------------------------------------------
// Kernel 2: sparse flash attention.
// Block: 256 threads = 8 waves; wave w owns rows [i0b + 16w, i0b + 16w + 15].
// Grid: (L/128, B) = (32, 4).
// ---------------------------------------------------------------------------
__global__ __launch_bounds__(256, 1)
void bb_attn(const float* __restrict__ qkv,
             const unsigned* __restrict__ fineBM,
             const unsigned* __restrict__ coarseBM,
             float* __restrict__ out) {
    const int b    = blockIdx.y;
    const int i0b  = blockIdx.x * 128;
    const int wave = threadIdx.x >> 5;
    const int lane = threadIdx.x & 31;
    const int nl   = lane & 15;     // lane-in-half: N index / A-row index
    const int hi   = lane >> 4;     // lane half
    const int i0w  = i0b + wave * 16;

    __shared__ _Float16 Klds[32 * KPITCH];          //  49,664 B (row major)
    __shared__ _Float16 Vlds[DMODEL * VPITCH];      //  61,440 B (transposed)
    __shared__ _Float16 Plds[8][16 * PPITCH];       //  10,240 B (per wave)
    __shared__ unsigned wgC[8];                     // union coarse row (block)
    __shared__ unsigned myC[8][8];                  // per-wave coarse row

    // ---- coarse bitmap rows into LDS -------------------------------------
    if (threadIdx.x < 64) {
        int w = threadIdx.x >> 3, q = threadIdx.x & 7;
        myC[w][q] = coarseBM[((i0b >> 4) + w) * 8 + q];
    }
    __syncthreads();
    if (threadIdx.x < 8) {
        unsigned u = 0;
        #pragma unroll
        for (int t = 0; t < 8; ++t) u |= myC[t][threadIdx.x];
        wgC[threadIdx.x] = u;
    }
    __syncthreads();

    // ---- Q fragments (A layout, 16x32 f16 per chunk) into VGPRs ----------
    // A layout: lane nl = row M; elems 0..7  = K (8*hi   .. 8*hi+7)
    //                            elems 8..15 = K (16+8*hi.. 16+8*hi+7)
    v16h qa[NKCHUNK];
    {
        const float* qrow = qkv + ((size_t)b * L_TOK + (i0w + nl)) * D3;
        #pragma unroll
        for (int c = 0; c < NKCHUNK; ++c) {
            const float4 x0 = *(const float4*)(qrow + 32 * c + 8 * hi);
            const float4 x1 = *(const float4*)(qrow + 32 * c + 8 * hi + 4);
            const float4 x2 = *(const float4*)(qrow + 32 * c + 16 + 8 * hi);
            const float4 x3 = *(const float4*)(qrow + 32 * c + 16 + 8 * hi + 4);
            v16h q;
            q[0]  = (_Float16)x0.x; q[1]  = (_Float16)x0.y;
            q[2]  = (_Float16)x0.z; q[3]  = (_Float16)x0.w;
            q[4]  = (_Float16)x1.x; q[5]  = (_Float16)x1.y;
            q[6]  = (_Float16)x1.z; q[7]  = (_Float16)x1.w;
            q[8]  = (_Float16)x2.x; q[9]  = (_Float16)x2.y;
            q[10] = (_Float16)x2.z; q[11] = (_Float16)x2.w;
            q[12] = (_Float16)x3.x; q[13] = (_Float16)x3.y;
            q[14] = (_Float16)x3.z; q[15] = (_Float16)x3.w;
            qa[c] = q;
        }
    }

    // ---- constant all-ones B fragment for WMMA row-sum -------------------
    v16h ones16;
    #pragma unroll
    for (int e = 0; e < 16; ++e) ones16[e] = (_Float16)1.0f;

    // ---- flash-attention state (C-fragment layout: VGPR v <-> rows v,v+8)
    v8f acc[NNCHUNK];
    const v8f vzero = {0.f,0.f,0.f,0.f,0.f,0.f,0.f,0.f};
    #pragma unroll
    for (int n = 0; n < NNCHUNK; ++n) acc[n] = vzero;
    float rowm[8], rsum[8];
    #pragma unroll
    for (int v = 0; v < 8; ++v) { rowm[v] = -1e30f; rsum[v] = 0.f; }

    const int jmax = i0b + 127;          // causal upper bound for this block
    for (int j0 = 0; j0 <= jmax; j0 += 32) {
        const int jbit = (j0 >> 4) & 31;
        const bool wgRand = ((wgC[j0 >> 9] >> jbit) & 3u) != 0;
        const bool wgAct  = (j0 < GLOBALC) || (j0 + 31 + WINDOW >= i0b) || wgRand;
        if (!wgAct) continue;            // block-uniform skip (no K/V load)

        // ---- cooperative K tile load: rows j0..j0+31, f16 row-major ------
        for (int idx = threadIdx.x; idx < 32 * 192; idx += 256) {
            int jj = idx / 192, fg = idx - jj * 192, f = fg * 4;
            const float4 x = *(const float4*)(qkv +
                ((size_t)b * L_TOK + (j0 + jj)) * D3 + DMODEL + f);
            h2 a = {(_Float16)x.x, (_Float16)x.y};
            h2 bb2 = {(_Float16)x.z, (_Float16)x.w};
            *(h2*)&Klds[jj * KPITCH + f]     = a;
            *(h2*)&Klds[jj * KPITCH + f + 2] = bb2;
        }
        // ---- cooperative V tile load, TRANSPOSED: Vlds[f][j] -------------
        for (int idx = threadIdx.x; idx < 16 * 192; idx += 256) {
            int jp = idx / 192, fg = idx - jp * 192, f = fg * 4, jj = jp * 2;
            const float4 x = *(const float4*)(qkv +
                ((size_t)b * L_TOK + (j0 + jj)) * D3 + 2 * DMODEL + f);
            const float4 y = *(const float4*)(qkv +
                ((size_t)b * L_TOK + (j0 + jj + 1)) * D3 + 2 * DMODEL + f);
            h2 e0 = {(_Float16)x.x, (_Float16)y.x};
            h2 e1 = {(_Float16)x.y, (_Float16)y.y};
            h2 e2 = {(_Float16)x.z, (_Float16)y.z};
            h2 e3 = {(_Float16)x.w, (_Float16)y.w};
            *(h2*)&Vlds[(f + 0) * VPITCH + jj] = e0;
            *(h2*)&Vlds[(f + 1) * VPITCH + jj] = e1;
            *(h2*)&Vlds[(f + 2) * VPITCH + jj] = e2;
            *(h2*)&Vlds[(f + 3) * VPITCH + jj] = e3;
        }
        __syncthreads();

        const bool myRand  = ((myC[wave][j0 >> 9] >> jbit) & 3u) != 0;
        const bool waveAct = (j0 <= i0w + 15) &&
                             ((j0 < GLOBALC) || (j0 + 31 + WINDOW >= i0w) || myRand);
        if (waveAct) {
            // ---- S = Q * K^T : two 16x16 f32 C tiles (cols 0-15, 16-31) --
            // Software-pipelined: next B fragment issued before each WMMA.
            v8f c0 = vzero, c1 = vzero;
            v16h kb0 = ld_v16h(&Klds[nl * KPITCH + 8 * hi]);   // c=0, tile0
            #pragma unroll
            for (int c = 0; c < NKCHUNK; ++c) {
                v16h kb1 = ld_v16h(&Klds[(nl + 16) * KPITCH + 32 * c + 8 * hi]);
                c0 = wmma_f16(qa[c], kb0, c0);
                if (c + 1 < NKCHUNK)
                    kb0 = ld_v16h(&Klds[nl * KPITCH + 32 * (c + 1) + 8 * hi]);
                c1 = wmma_f16(qa[c], kb1, c1);
            }

            // ---- mask + online softmax (max via permlane16 butterfly) ----
            float fac[8];
            #pragma unroll
            for (int v = 0; v < 8; ++v) {
                const int irow = i0w + v + 8 * hi;
                const int ja = j0 + nl;        // col of c0 element
                const int jb = ja + 16;        // col of c1 element
                unsigned wbits = 0u;
                if (myRand) wbits = fineBM[irow * FINE_WORDS_PER_ROW + (j0 >> 5)];
                float s0 = c0[v] * SCALE_Q;
                float s1 = c1[v] * SCALE_Q;
                const bool a0 = (ja <= irow) &&
                    ((ja + WINDOW >= irow) || (ja < GLOBALC) || ((wbits >> nl) & 1u));
                const bool a1 = (jb <= irow) &&
                    ((jb + WINDOW >= irow) || (jb < GLOBALC) || ((wbits >> (nl + 16)) & 1u));
                s0 = a0 ? s0 : -1e30f;
                s1 = a1 ? s1 : -1e30f;
                // per-row max across the 16 lanes of this half-wave
                const float t  = half16_max(fmaxf(s0, s1));
                const float nm = fmaxf(rowm[v], t);
                fac[v] = __expf(rowm[v] - nm);
                const float p0 = (s0 < -1e29f) ? 0.f : __expf(s0 - nm);
                const float p1 = (s1 < -1e29f) ? 0.f : __expf(s1 - nm);
                rowm[v] = nm;
                // stage P (row-major 16x32 f16, per-wave region)
                const int r = v + 8 * hi;
                Plds[wave][r * PPITCH + nl]      = (_Float16)p0;
                Plds[wave][r * PPITCH + 16 + nl] = (_Float16)p1;
            }
            asm volatile("s_wait_dscnt 0" ::: "memory");

            // ---- P as A fragment (16x32, K = 32 j-columns) ---------------
            v16h pa;
            {
                const _Float16* pp = &Plds[wave][nl * PPITCH];
                v8h lo  = *(const v8h*)(pp + 8 * hi);
                v8h hi8 = *(const v8h*)(pp + 16 + 8 * hi);
                pa = __builtin_shufflevector(lo, hi8,
                        0,1,2,3,4,5,6,7,8,9,10,11,12,13,14,15);
            }

            // ---- row sums via WMMA against all-ones B (consumed after the
            // PV loop so the WMMA->VALU hazard is covered by 48 WMMAs) -----
            const v8f csum = wmma_f16(pa, ones16, vzero);

            // ---- O = diag(fac)*O + P * V  (pipelined B prefetch) ---------
            v16h vb = ld_v16h(&Vlds[nl * VPITCH + 8 * hi]);    // n=0
            #pragma unroll
            for (int n = 0; n < NNCHUNK; ++n) {
                v8f a = acc[n];
                #pragma unroll
                for (int v = 0; v < 8; ++v) a[v] = a[v] * fac[v];
                const v16h cur = vb;
                if (n + 1 < NNCHUNK)
                    vb = ld_v16h(&Vlds[(16 * (n + 1) + nl) * VPITCH + 8 * hi]);
                acc[n] = wmma_f16(pa, cur, a);
            }

            #pragma unroll
            for (int v = 0; v < 8; ++v)
                rsum[v] = rsum[v] * fac[v] + csum[v];
        }
        __syncthreads();
    }

    // ---- epilogue: out = acc / rowsum ------------------------------------
    float inv[8];
    #pragma unroll
    for (int v = 0; v < 8; ++v) inv[v] = 1.0f / rsum[v];
    #pragma unroll
    for (int n = 0; n < NNCHUNK; ++n) {
        #pragma unroll
        for (int v = 0; v < 8; ++v) {
            const int irow = i0w + v + 8 * hi;
            out[((size_t)b * L_TOK + irow) * DMODEL + 16 * n + nl] = acc[n][v] * inv[v];
        }
    }
}

// ---------------------------------------------------------------------------
extern "C" void kernel_launch(void* const* d_in, const int* in_sizes, int n_in,
                              void* d_out, int out_size, void* d_ws, size_t ws_size,
                              hipStream_t stream) {
    const float* qkv = (const float*)d_in[0];
    const int*   pos0 = (const int*)d_in[1];
    const int*   pos1 = (const int*)d_in[2];
    float* outp = (float*)d_out;

    unsigned* fineBM   = (unsigned*)d_ws;
    unsigned* coarseBM = fineBM + FINE_WORDS;
    const size_t bmBytes = (size_t)(FINE_WORDS + COARSE_WORDS) * sizeof(unsigned);

    hipMemsetAsync(d_ws, 0, bmBytes, stream);

    const int nrand = in_sizes[1];
    bb_scatter<<<(nrand + 127) / 128, 128, 0, stream>>>(pos0, pos1, fineBM, coarseBM, nrand);

    bb_attn<<<dim3(L_TOK / 128, NBATCH), 256, 0, stream>>>(qkv, fineBM, coarseBM, outp);
}
